// GCN_EL_H_38714835206183
// MI455X (gfx1250) — matrix operand order repaired and checked by reading, not compiled, not run on hardware
//
#include <hip/hip_runtime.h>
#include <hip/hip_bf16.h>

// ---------------------------------------------------------------------------
// Problem constants (match reference)
// ---------------------------------------------------------------------------
constexpr int NN   = 8192;   // nodes
constexpr int E2   = 128;    // 2*E (node embedding width)
constexpr int HH   = 128;    // hidden
constexpr int CC   = 512;    // communities
constexpr int MEX  = 2048;   // duplicated multi-community entries

typedef __attribute__((ext_vector_type(16))) _Float16 v16h;
typedef __attribute__((ext_vector_type(8)))  _Float16 v8h;
typedef __attribute__((ext_vector_type(8)))  float    v8f;
typedef __attribute__((ext_vector_type(4)))  float    v4f;

// ---------------------------------------------------------------------------
// WMMA fragment loaders (wave32 layouts per CDNA5 ISA 7.12.2)
//
// A (16x32 f16): lane L -> M = L&15 ; elems 0..7  at K = (L<16?0:8)  + e
//                                      elems 8..15 at K = (L<16?16:24)+ e-8
// B (32x16 f16): lane L -> N = L&15 ; same K mapping as A (elems 0..7 at (L<16?0:8)+e, 8..15 at (L<16?16:24)+e-8)
// Both are two contiguous 16-byte loads per lane from a row-major source.
// ---------------------------------------------------------------------------
__device__ __forceinline__ v16h load_frag_A(const _Float16* __restrict__ M, int ld,
                                            int rowbase, int kbase, int lane) {
  const int m  = lane & 15;
  const int hi = lane >> 4;
  const _Float16* p = M + (size_t)(rowbase + m) * ld + kbase + hi * 8;
  v8h lo = *(const v8h*)p;
  v8h hh = *(const v8h*)(p + 16);
  v16h r;
#pragma unroll
  for (int e = 0; e < 8; ++e) { r[e] = lo[e]; r[e + 8] = hh[e]; }
  return r;
}

__device__ __forceinline__ v16h load_frag_B(const _Float16* __restrict__ M, int ld,
                                            int nbase, int kbase, int lane) {
  const int n  = lane & 15;
  const int ko = (lane >> 4) * 8;
  const _Float16* p = M + (size_t)(nbase + n) * ld + kbase + ko;
  v8h lo = *(const v8h*)p;
  v8h hh = *(const v8h*)(p + 16);
  v16h r;
#pragma unroll
  for (int e = 0; e < 8; ++e) { r[e] = lo[e]; r[e + 8] = hh[e]; }
  return r;
}

// ---------------------------------------------------------------------------
// Fused edge-learning + masked graph convolution (the hot kernel, WMMA).
//   out[i,:] = relu( (sum_j sigmoid(z_i.z_j/sqrt(128))*adj_ij * hw[j,:]) / deg_i + b )
// One block = 16 output rows; 8 waves split the j dimension in 32-col tiles.
// adj streams once from HBM (non-temporal); z/hwT stay L2-resident.
// ---------------------------------------------------------------------------
__global__ __launch_bounds__(256)
void k_conv_fused(const _Float16* __restrict__ Z,    // [NN,128] f16
                  const _Float16* __restrict__ HWT,  // [128,NN] f16 (hW transposed)
                  const float*    __restrict__ ADJ,  // [NN,NN]
                  const float*    __restrict__ bias, // [128]
                  float*          __restrict__ OUT,  // [NN,128]
                  int add_mode)
{
  __shared__ float    red[16][HH];          // cross-wave accumulator
  __shared__ float    degS[16];             // per-row degree
  __shared__ _Float16 Slds[8][16][32];      // per-wave score staging (C->A transpose)

  const int tid  = threadIdx.x;
  const int lane = tid & 31;
  const int wave = tid >> 5;
  const int i0   = blockIdx.x * 16;
  const int m    = lane & 15;
  const int hi   = lane >> 4;

  for (int k = tid; k < 16 * HH; k += 256) (&red[0][0])[k] = 0.f;
  if (tid < 16) degS[tid] = 0.f;
  __syncthreads();

  // A-fragments of z rows i0..i0+15 over K=128 (4 chunks of 32), kept in VGPRs
  v16h aZ[4];
#pragma unroll
  for (int kc = 0; kc < 4; ++kc) aZ[kc] = load_frag_A(Z, E2, i0, kc * 32, lane);

  v8f acc[8] = {};           // out accumulator: 8 tiles of 16 H-columns
  float degacc = 0.f;
  const float* adjrow = ADJ + (size_t)(i0 + m) * NN;
  const float inv_sqrt = 0.088388347648318447f;   // 1/sqrt(128)

  for (int jb = wave * 32; jb < NN; jb += 256) {
    // ---- score tile S[16 x 32] = z_i (16x128) . z_j^T ----
    v8f S0 = {}, S1 = {};
#pragma unroll
    for (int kc = 0; kc < 4; ++kc) {
      v16h b0 = load_frag_B(Z, E2, jb,      kc * 32, lane);
      v16h b1 = load_frag_B(Z, E2, jb + 16, kc * 32, lane);
      S0 = __builtin_amdgcn_wmma_f32_16x16x32_f16(false, aZ[kc], false, b0,
                                                  (short)0, S0, false, false);
      S1 = __builtin_amdgcn_wmma_f32_16x16x32_f16(false, aZ[kc], false, b1,
                                                  (short)0, S1, false, false);
    }

    // sigmoid, scatter C-layout -> LDS [row][k] (f16)
#pragma unroll
    for (int r = 0; r < 8; ++r) {
      int row = r + 8 * hi;
      float s0 = 1.f / (1.f + __expf(-S0[r] * inv_sqrt));
      float s1 = 1.f / (1.f + __expf(-S1[r] * inv_sqrt));
      Slds[wave][row][m]      = (_Float16)s0;
      Slds[wave][row][16 + m] = (_Float16)s1;
    }
    asm volatile("s_wait_dscnt 0x0" ::: "memory");   // same-wave LDS RAW

    // re-read as A-fragment, masking with adj and accumulating degree.
    // adj is a 256MB single-use stream: load non-temporal so it never evicts
    // the L2-resident z / hwT working set.
    const float* arow = adjrow + jb;
    v4f a0 = __builtin_nontemporal_load((const v4f*)(arow + hi * 8));
    v4f a1 = __builtin_nontemporal_load((const v4f*)(arow + hi * 8 + 4));
    v4f a2 = __builtin_nontemporal_load((const v4f*)(arow + 16 + hi * 8));
    v4f a3 = __builtin_nontemporal_load((const v4f*)(arow + 16 + hi * 8 + 4));
    v16h SA;
#pragma unroll
    for (int e = 0; e < 8; ++e) {
      float av = (e < 4) ? a0[e] : a1[e - 4];
      float p  = (float)Slds[wave][m][hi * 8 + e] * av;
      degacc += p;
      SA[e] = (_Float16)p;
    }
#pragma unroll
    for (int e = 0; e < 8; ++e) {
      float av = (e < 4) ? a2[e] : a3[e - 4];
      float p  = (float)Slds[wave][m][16 + hi * 8 + e] * av;
      degacc += p;
      SA[8 + e] = (_Float16)p;
    }

    // ---- out[16 x 128] += S(16x32) . hw[jb..jb+31, 0..127] ----
#pragma unroll
    for (int t = 0; t < 8; ++t) {
      v16h bh = load_frag_B(HWT, NN, t * 16, jb, lane);  // B[k][n] = hwT[n0+n][jb+k]
      acc[t] = __builtin_amdgcn_wmma_f32_16x16x32_f16(false, SA, false, bh,
                                                      (short)0, acc[t], false, false);
    }
  }

  // cross-wave reduction (C layout: lane -> col L&15, rows r+8*hi)
#pragma unroll
  for (int t = 0; t < 8; ++t)
#pragma unroll
    for (int r = 0; r < 8; ++r)
      atomicAdd(&red[r + 8 * hi][t * 16 + m], acc[t][r]);
  atomicAdd(&degS[m], degacc);
  __syncthreads();

  for (int k = tid; k < 16 * HH; k += 256) {
    int row = k >> 7, col = k & 127;
    float d = degS[row]; d = d < 1.f ? 1.f : d;
    float v = red[row][col] / d + bias[col];
    v = v > 0.f ? v : 0.f;
    size_t idx = (size_t)(i0 + row) * HH + col;
    if (add_mode) OUT[idx] += v; else OUT[idx] = v;
  }
}

// ---------------------------------------------------------------------------
// Small dense kernels (all <0.5 GFLOP total; VALU is fine, data L2-resident)
// ---------------------------------------------------------------------------
__global__ void k_edge_embed(const float* __restrict__ ea, const float* __restrict__ W,
                             const float* __restrict__ b, float* __restrict__ out) {
  int idx = blockIdx.x * 256 + threadIdx.x;      // NN*64 threads
  int n = idx >> 6, c = idx & 63;
  float s = b[c];
#pragma unroll
  for (int k = 0; k < 8; ++k) s += ea[n * 8 + k] * W[k * 64 + c];
  out[(size_t)n * 64 + c] = fmaxf(s, 0.f);
}

__global__ void k_node_embed(const float* __restrict__ x,
                             const float* __restrict__ W1, const float* __restrict__ b1,
                             const float* __restrict__ W2, const float* __restrict__ b2,
                             float* __restrict__ t) {
  int idx = blockIdx.x * 256 + threadIdx.x;      // NN*128 threads
  int n = idx >> 7, c = idx & 127;
  float s;
  if (c < 64) {
    s = b1[c];
#pragma unroll
    for (int k = 0; k < 8; ++k) s += x[n * 20 + k] * W1[k * 64 + c];
  } else {
    int c2 = c - 64;
    s = b2[c2];
#pragma unroll
    for (int k = 0; k < 12; ++k) s += x[n * 20 + 8 + k] * W2[k * 64 + c2];
  }
  t[(size_t)n * 128 + c] = fmaxf(s, 0.f);
}

__global__ void k_gemm128_relu(const float* __restrict__ t, const float* __restrict__ W,
                               const float* __restrict__ b, float* __restrict__ h) {
  int idx = blockIdx.x * 256 + threadIdx.x;      // NN*128 threads
  int n = idx >> 7, c = idx & 127;
  float s = b[c];
  for (int k = 0; k < 128; ++k) s += t[(size_t)n * 128 + k] * W[k * 128 + c];
  h[(size_t)n * 128 + c] = fmaxf(s, 0.f);
}

// z = relu(h@Wn + bn) + relu(ea@We + be), stored f16 for WMMA score compute
__global__ void k_z(const float* __restrict__ h, const float* __restrict__ Wn,
                    const float* __restrict__ bn, const float* __restrict__ ea,
                    const float* __restrict__ We, const float* __restrict__ be,
                    _Float16* __restrict__ z) {
  int idx = blockIdx.x * 256 + threadIdx.x;      // NN*128 threads
  int n = idx >> 7, c = idx & 127;
  float q = bn[c];
  for (int k = 0; k < 128; ++k) q += h[(size_t)n * 128 + k] * Wn[k * 128 + c];
  q = fmaxf(q, 0.f);
  float g = be[c];
  for (int k = 0; k < 64; ++k) g += ea[(size_t)n * 64 + k] * We[k * 128 + c];
  g = fmaxf(g, 0.f);
  z[(size_t)n * 128 + c] = (_Float16)(q + g);
}

// hwT[c][n] = (h @ W)[n][c], transposed + f16 so conv B-fragments load contiguously
__global__ void k_hwT(const float* __restrict__ h, const float* __restrict__ W,
                      _Float16* __restrict__ hwT) {
  int idx = blockIdx.x * 256 + threadIdx.x;      // NN*128 threads, n fastest
  int n = idx & (NN - 1), c = idx >> 13;
  float s = 0.f;
  for (int k = 0; k < 128; ++k) s += h[(size_t)n * 128 + k] * W[k * 128 + c];
  hwT[(size_t)c * NN + n] = (_Float16)s;
}

// ---------------------------------------------------------------------------
// Community pooling: h >= 0 (sum of relus) so float max == int max on bits,
// and the 0-init matches the reference's empty-segment fill.
// p layout: [C][256], cols 0..127 = mean-sum, cols 128..255 = max.
// ---------------------------------------------------------------------------
__global__ void k_cnt(const int* __restrict__ community, const int* __restrict__ mci,
                      float* __restrict__ cnt) {
  int e = blockIdx.x * 256 + threadIdx.x;        // NN+MEX threads
  int seg = (e < NN) ? community[e] : mci[e - NN];
  atomicAdd(&cnt[seg], 1.f);
}

__global__ void k_pool_acc(const float* __restrict__ h, const int* __restrict__ community,
                           const int* __restrict__ mcn, const int* __restrict__ mci,
                           float* __restrict__ p) {
  int idx = blockIdx.x * 256 + threadIdx.x;      // (NN+MEX)*128 threads
  int e = idx >> 7, f = idx & 127;
  int node, seg;
  if (e < NN) { node = e; seg = community[e]; }
  else        { node = mcn[e - NN]; seg = mci[e - NN]; }
  float v = h[(size_t)node * 128 + f];
  atomicAdd(&p[(size_t)seg * 256 + f], v);
  atomicMax((int*)&p[(size_t)seg * 256 + 128 + f], __float_as_int(v));
}

__global__ void k_pool_fin(float* __restrict__ p, const float* __restrict__ cnt) {
  int idx = blockIdx.x * 256 + threadIdx.x;      // C*128 threads
  int c = idx >> 7, f = idx & 127;
  p[(size_t)c * 256 + f] /= fmaxf(cnt[c], 1.f);
}

// head: s = relu((p1+p2) @ W1 + b1) ; out = s @ W2 + b2
__global__ __launch_bounds__(128)
void k_head(const float* __restrict__ p1, const float* __restrict__ p2,
            const float* __restrict__ W1, const float* __restrict__ b1,
            const float* __restrict__ W2, const float* __restrict__ b2,
            float* __restrict__ out) {
  __shared__ float sm[128];
  int c = blockIdx.x, t = threadIdx.x;
  float s = b1[t];
  for (int k = 0; k < 256; ++k)
    s += (p1[(size_t)c * 256 + k] + p2[(size_t)c * 256 + k]) * W1[k * 128 + t];
  s = fmaxf(s, 0.f);
  sm[t] = s * W2[t];
  __syncthreads();
  for (int off = 64; off > 0; off >>= 1) {
    if (t < off) sm[t] += sm[t + off];
    __syncthreads();
  }
  if (t == 0) out[c] = sm[0] + b2[0];
}

// ---------------------------------------------------------------------------
// Orchestration
// ---------------------------------------------------------------------------
extern "C" void kernel_launch(void* const* d_in, const int* in_sizes, int n_in,
                              void* d_out, int out_size, void* d_ws, size_t ws_size,
                              hipStream_t stream) {
  (void)in_sizes; (void)n_in; (void)out_size; (void)ws_size;

  const float* x      = (const float*)d_in[0];
  const float* eaIraw = (const float*)d_in[1];
  const float* eaAraw = (const float*)d_in[2];
  const float* adjI   = (const float*)d_in[3];
  const float* adjA   = (const float*)d_in[4];
  const int*   comm   = (const int*)d_in[5];
  const int*   mcn    = (const int*)d_in[6];
  const int*   mci    = (const int*)d_in[7];
  // params, dict order, W then b each
  const float* Wee1=(const float*)d_in[8],  *bee1=(const float*)d_in[9];
  const float* Wee2=(const float*)d_in[10], *bee2=(const float*)d_in[11];
  const float* Wem1=(const float*)d_in[12], *bem1=(const float*)d_in[13];
  const float* Wem2=(const float*)d_in[14], *bem2=(const float*)d_in[15];
  const float* Wem3=(const float*)d_in[16], *bem3=(const float*)d_in[17];
  const float* Wn1 =(const float*)d_in[18], *bn1 =(const float*)d_in[19];
  const float* We1 =(const float*)d_in[20], *be1 =(const float*)d_in[21];
  const float* Wn2 =(const float*)d_in[22], *bn2 =(const float*)d_in[23];
  const float* We2 =(const float*)d_in[24], *be2 =(const float*)d_in[25];
  const float* Wci1=(const float*)d_in[26], *bci1=(const float*)d_in[27];
  const float* Wca1=(const float*)d_in[28], *bca1=(const float*)d_in[29];
  const float* Wci2=(const float*)d_in[30], *bci2=(const float*)d_in[31];
  const float* Wca2=(const float*)d_in[32], *bca2=(const float*)d_in[33];
  const float* Wl1 =(const float*)d_in[34], *bl1 =(const float*)d_in[35];
  const float* Wl2 =(const float*)d_in[36], *bl2 =(const float*)d_in[37];

  char* ws = (char*)d_ws;
  const size_t MB = 1024 * 1024;
  float*    h0   = (float*)(ws + 0 * MB);       // [NN,128]  4MB
  float*    h1   = (float*)(ws + 4 * MB);       // [NN,128]  4MB
  float*    h2   = (float*)(ws + 8 * MB);       // [NN,128]  4MB
  float*    tbuf = (float*)(ws + 12 * MB);      // [NN,128]  4MB
  float*    eI   = (float*)(ws + 16 * MB);      // [NN,64]   2MB
  float*    eA   = (float*)(ws + 18 * MB);      // [NN,64]   2MB
  _Float16* zI   = (_Float16*)(ws + 20 * MB);   // [NN,128]  2MB
  _Float16* zA   = (_Float16*)(ws + 22 * MB);   // [NN,128]  2MB
  _Float16* hwT  = (_Float16*)(ws + 24 * MB);   // [128,NN]  2MB
  float*    p1   = (float*)(ws + 26 * MB);      // [C,256]   512KB
  float*    p2   = (float*)(ws + 26 * MB + 512 * 1024);
  float*    cnt  = (float*)(ws + 27 * MB);      // [C]

  dim3 blk(256);
  // embeddings
  k_edge_embed <<<NN*64/256,  blk, 0, stream>>>(eaIraw, Wee1, bee1, eI);
  k_edge_embed <<<NN*64/256,  blk, 0, stream>>>(eaAraw, Wee2, bee2, eA);
  k_node_embed <<<NN*128/256, blk, 0, stream>>>(x, Wem1, bem1, Wem2, bem2, tbuf);
  k_gemm128_relu<<<NN*128/256, blk, 0, stream>>>(tbuf, Wem3, bem3, h0);
  // edge-learning z (masks are computed once from h0 and reused by both layers)
  k_z<<<NN*128/256, blk, 0, stream>>>(h0, Wn1, bn1, eI, We1, be1, zI);
  k_z<<<NN*128/256, blk, 0, stream>>>(h0, Wn2, bn2, eA, We2, be2, zA);

  hipMemsetAsync(p1, 0, (size_t)CC * 256 * sizeof(float), stream);
  hipMemsetAsync(p2, 0, (size_t)CC * 256 * sizeof(float), stream);
  hipMemsetAsync(cnt, 0, (size_t)CC * sizeof(float), stream);

  // layer 1
  k_hwT<<<NN*128/256, blk, 0, stream>>>(h0, Wci1, hwT);
  k_conv_fused<<<NN/16, blk, 0, stream>>>(zI, hwT, adjI, bci1, h1, 0);
  k_hwT<<<NN*128/256, blk, 0, stream>>>(h0, Wca1, hwT);
  k_conv_fused<<<NN/16, blk, 0, stream>>>(zA, hwT, adjA, bca1, h1, 1);

  k_cnt     <<<(NN+MEX)/256,      blk, 0, stream>>>(comm, mci, cnt);
  k_pool_acc<<<(NN+MEX)*128/256,  blk, 0, stream>>>(h1, comm, mcn, mci, p1);
  k_pool_fin<<<CC*128/256,        blk, 0, stream>>>(p1, cnt);

  // layer 2 (same masks, new features)
  k_hwT<<<NN*128/256, blk, 0, stream>>>(h1, Wci2, hwT);
  k_conv_fused<<<NN/16, blk, 0, stream>>>(zI, hwT, adjI, bci2, h2, 0);
  k_hwT<<<NN*128/256, blk, 0, stream>>>(h1, Wca2, hwT);
  k_conv_fused<<<NN/16, blk, 0, stream>>>(zA, hwT, adjA, bca2, h2, 1);

  k_pool_acc<<<(NN+MEX)*128/256,  blk, 0, stream>>>(h2, comm, mcn, mci, p2);
  k_pool_fin<<<CC*128/256,        blk, 0, stream>>>(p2, cnt);

  k_head<<<CC, 128, 0, stream>>>(p1, p2, Wl1, bl1, Wl2, bl2, (float*)d_out);
}
